// GCNModel_12008728560135
// MI455X (gfx1250) — matrix operand order repaired and checked by reading, not compiled, not run on hardware
//
#include <hip/hip_runtime.h>
#include <hip/hip_bf16.h>

typedef __attribute__((ext_vector_type(16))) __bf16 v16bf;
typedef __attribute__((ext_vector_type(8)))  __bf16 v8bf;
typedef __attribute__((ext_vector_type(8)))  float  v8f;

#define COUT 128  // both layers produce 128 features

// ---------------- small prep kernels ----------------

// Wt[c*K + k] = (bf16) W[k*NC + c]   (transpose to [N][K] so B-fragments are contiguous in K)
__global__ void k_transpose_w(const float* __restrict__ W, __bf16* __restrict__ Wt,
                              int K, int NC) {
    int c = blockIdx.x;      // output column (0..NC-1)
    int k = threadIdx.x;     // 0..K-1
    Wt[(size_t)c * K + k] = (__bf16)W[(size_t)k * NC + c];
}

__global__ void k_fill1(float* __restrict__ p, int n) {
    int i = blockIdx.x * blockDim.x + threadIdx.x;
    if (i < n) p[i] = 1.0f;   // self-loop contributes 1 to every degree
}

__global__ void k_deg(const int* __restrict__ col, float* __restrict__ deg, int E) {
    int i = blockIdx.x * blockDim.x + threadIdx.x;
    if (i < E) unsafeAtomicAdd(&deg[col[i]], 1.0f);
}

__global__ void k_rsqrt(float* __restrict__ p, int n) {
    int i = blockIdx.x * blockDim.x + threadIdx.x;
    if (i < n) p[i] = rsqrtf(p[i]);   // deg >= 1 always (self-loops)
}

// ---------------- bf16 WMMA GEMM: Y[M,128] = X[M,K] @ W[K,128] ----------------
// 256 threads = 8 waves; each wave computes a 16-row strip x 128 cols.
// K is 256 (layer 1) or 128 (layer 2), always a multiple of 32.
__global__ __launch_bounds__(256) void k_gemm(const float* __restrict__ X,
                                              const __bf16* __restrict__ Wt, // [128][K] bf16
                                              float* __restrict__ Y,
                                              int M, int K) {
    const int wave = threadIdx.x >> 5;
    const int lane = threadIdx.x & 31;
    const int lh   = lane >> 4;   // lane half: 0 or 1
    const int ll   = lane & 15;
    const int m0   = blockIdx.x * 128 + wave * 16;

    int rowm = m0 + ll;
    if (rowm > M - 1) rowm = M - 1;           // clamp (stores are guarded)
    const float* xrow = X + (size_t)rowm * K;
    const int khalf = lh * 16;                // A layout: lanes 0-15 hold K {0..15}

    // B fragment base: lane ll holds column ll of the tile, K range per lane half.
    const __bf16* wbase = Wt + (size_t)ll * K + lh * 16;

    v8f acc[8] = {};  // 8 N-tiles of 16, f32 accumulators

    for (int k0 = 0; k0 < K; k0 += 32) {
        // A fragment: 16 bf16 per lane, contiguous k = 16*lh + s (same order as B)
        v16bf a;
        const float* p0 = xrow + k0 + khalf;
        const float* p1 = p0 + 8;
#pragma unroll
        for (int i = 0; i < 8; ++i) {
            a[i]     = (__bf16)p0[i];
            a[i + 8] = (__bf16)p1[i];
        }
#pragma unroll
        for (int t = 0; t < 8; ++t) {
            const __bf16* wp = wbase + (size_t)t * 16 * K + k0;
            v8bf blo = *(const v8bf*)(wp);
            v8bf bhi = *(const v8bf*)(wp + 8);
            v16bf b  = __builtin_shufflevector(blo, bhi,
                        0,1,2,3,4,5,6,7,8,9,10,11,12,13,14,15);
            acc[t] = __builtin_amdgcn_wmma_f32_16x16x32_bf16(
                        false, a, false, b, (short)0, acc[t], false, false);
        }
    }

    // C layout: VGPR r, lanes 0-15 -> M=r, N=lane; lanes 16-31 -> M=r+8, N=lane-16
#pragma unroll
    for (int t = 0; t < 8; ++t) {
#pragma unroll
        for (int r = 0; r < 8; ++r) {
            int m = m0 + r + lh * 8;
            if (m < M) Y[(size_t)m * COUT + t * 16 + ll] = acc[t][r];
        }
    }
}

// ---------------- aggregation kernels ----------------

// out = xw * dinv[i]^2  (self-loop message; also zero-initializes poisoned out)
__global__ void k_selfinit(const float* __restrict__ xw, const float* __restrict__ dinv,
                           float* __restrict__ out, int n /* = N*128 */) {
    int idx = blockIdx.x * blockDim.x + threadIdx.x;
    if (idx < n) {
        float d = dinv[idx >> 7];
        out[idx] = xw[idx] * d * d;
    }
}

// one wave32 per edge: lane covers 4 consecutive floats of the 128-wide feature row
__global__ __launch_bounds__(256) void k_scatter(const float* __restrict__ xw,
                                                 const float* __restrict__ dinv,
                                                 const int* __restrict__ row,
                                                 const int* __restrict__ col,
                                                 float* __restrict__ out, int E) {
    int w    = (blockIdx.x * blockDim.x + threadIdx.x) >> 5;
    int lane = threadIdx.x & 31;
    if (w >= E) return;
    int r = row[w];
    int c = col[w];
    float nrm = dinv[r] * dinv[c];
    const float4 v = *(const float4*)(xw + (size_t)r * COUT + lane * 4);
    float* op = out + (size_t)c * COUT + lane * 4;
    unsafeAtomicAdd(op + 0, v.x * nrm);
    unsafeAtomicAdd(op + 1, v.y * nrm);
    unsafeAtomicAdd(op + 2, v.z * nrm);
    unsafeAtomicAdd(op + 3, v.w * nrm);
}

__global__ void k_bias_relu(float* __restrict__ h, const float* __restrict__ b, int n) {
    int idx = blockIdx.x * blockDim.x + threadIdx.x;
    if (idx < n) {
        float v = h[idx] + b[idx & (COUT - 1)];
        h[idx] = v > 0.0f ? v : 0.0f;
    }
}

__global__ void k_bias(float* __restrict__ o, const float* __restrict__ b, int n) {
    int idx = blockIdx.x * blockDim.x + threadIdx.x;
    if (idx < n) o[idx] += b[idx & (COUT - 1)];
}

// ---------------- launcher ----------------

static inline size_t align256(size_t s) { return (s + 255) & ~(size_t)255; }

extern "C" void kernel_launch(void* const* d_in, const int* in_sizes, int n_in,
                              void* d_out, int out_size, void* d_ws, size_t ws_size,
                              hipStream_t stream) {
    const float* x  = (const float*)d_in[0];
    const int*   ei = (const int*)d_in[1];
    const float* W1 = (const float*)d_in[2];
    const float* b1 = (const float*)d_in[3];
    const float* W2 = (const float*)d_in[4];
    const float* b2 = (const float*)d_in[5];
    float* out = (float*)d_out;

    const int CIN = 256;
    const int N = in_sizes[0] / CIN;     // 100000
    const int E = in_sizes[1] / 2;       // 1600000
    const int* row = ei;
    const int* col = ei + E;

    // workspace carve-out
    char* ws = (char*)d_ws;
    size_t szNC = align256((size_t)N * COUT * sizeof(float)); // 51.2 MB
    float*  A    = (float*)(ws);                  // xw buffer (both layers)
    float*  B    = (float*)(ws + szNC);           // hidden buffer
    float*  dinv = (float*)(ws + 2 * szNC);       // degree -> dinv
    __bf16* Wt1  = (__bf16*)(ws + 2 * szNC + align256((size_t)N * sizeof(float)));
    __bf16* Wt2  = (__bf16*)((char*)Wt1 + align256((size_t)COUT * CIN * sizeof(__bf16)));

    const int NE = N * COUT;             // 12.8M elements

    // weight prep (bf16, transposed to [N][K])
    k_transpose_w<<<COUT, CIN,  0, stream>>>(W1, Wt1, CIN,  COUT);
    k_transpose_w<<<COUT, COUT, 0, stream>>>(W2, Wt2, COUT, COUT);

    // degree / normalization (shared by both layers)
    k_fill1 <<<(N + 255) / 256, 256, 0, stream>>>(dinv, N);
    k_deg   <<<(E + 255) / 256, 256, 0, stream>>>(col, dinv, E);
    k_rsqrt <<<(N + 255) / 256, 256, 0, stream>>>(dinv, N);

    // ---- layer 1 ----
    k_gemm     <<<(N + 127) / 128, 256, 0, stream>>>(x, Wt1, A, N, CIN);
    k_selfinit <<<(NE + 255) / 256, 256, 0, stream>>>(A, dinv, B, NE);
    k_scatter  <<<(E + 7) / 8, 256, 0, stream>>>(A, dinv, row, col, B, E);
    k_bias_relu<<<(NE + 255) / 256, 256, 0, stream>>>(B, b1, NE);

    // ---- layer 2 ----
    k_gemm     <<<(N + 127) / 128, 256, 0, stream>>>(B, Wt2, A, N, COUT);
    k_selfinit <<<(NE + 255) / 256, 256, 0, stream>>>(A, dinv, out, NE);
    k_scatter  <<<(E + 7) / 8, 256, 0, stream>>>(A, dinv, row, col, out, E);
    k_bias     <<<(NE + 255) / 256, 256, 0, stream>>>(out, b2, NE);
}
